// CausalSelfAttention_77807627534759
// MI455X (gfx1250) — matrix operand order repaired and checked by reading, not compiled, not run on hardware
//
#include <hip/hip_runtime.h>
#include <hip/hip_bf16.h>
#include <math.h>

// ---------------------------------------------------------------------------
// CDNA5 (gfx1250) causal self-attention forward, f16 WMMA / f32 accumulate:
//   0) one-shot fp32 -> f16 conversion of x, w_qkv, w_proj
//   1) qkv16 = x16 @ w_qkv16^T      (WMMA GEMM, f16 out, async LDS staging)
//   2) attn16 = flash-attention      (WMMA QK^T and PV, online softmax, causal)
//   3) out    = attn16 @ w_proj16^T  (WMMA GEMM, f32 out)
// B=2, T=2048, E=1024, H=16, D=64
// ---------------------------------------------------------------------------

typedef __attribute__((ext_vector_type(16))) _Float16 v16h;
typedef __attribute__((ext_vector_type(8)))  _Float16 h8;
typedef __attribute__((ext_vector_type(8)))  float    v8f;

#define CB  2
#define CT  2048
#define CE  1024
#define C3E 3072
#define CH  16
#define CD  64

// ---- async global -> LDS copy (CDNA5 ISA) ------------
// GV mode: per-lane LDS byte address in VDST, 64-bit global address in VADDR.
// INST_OFFSET is added to BOTH the LDS and global addresses.
#define ASYNC_LD_B128(ldsAddr, gAddr, OFF)                                   \
  asm volatile("global_load_async_to_lds_b128 %0, %1, off offset:" #OFF      \
               :: "v"(ldsAddr), "v"(gAddr) : "memory")
#define WAIT_ASYNC() asm volatile("s_wait_asynccnt 0x0" ::: "memory")

// Raw LDS byte offset of a pointer into __shared__ (low 32 bits of the address).
__device__ __forceinline__ unsigned lds_off(const void* p) {
  return (unsigned)(unsigned long long)p;
}

// ---- WMMA fragment loaders (layouts per CDNA5 ISA §7.12.2) ------
// A-operand 16x32 f16: lane = M row (lane&15); K slots interleaved:
//   lane-half 0: k = {0..7, 16..23}; lane-half 1: k = {8..15, 24..31}
// Works on any address space (LDS tile or global row-major matrix).
__device__ __forceinline__ v16h frag_ldA(const _Float16* tile, int ld) {
  const int lane = threadIdx.x & 31;
  const int row  = lane & 15;
  const int koff = (lane >> 4) << 3;          // 0 or 8
  const _Float16* p = tile + row * ld;
  h8 lo = *reinterpret_cast<const h8*>(p + koff);
  h8 hi = *reinterpret_cast<const h8*>(p + 16 + koff);
  return __builtin_shufflevector(lo, hi, 0,1,2,3,4,5,6,7,8,9,10,11,12,13,14,15);
}

// B-operand 32x16 f16 (stored as Bt[n][k], k contiguous): lane = N column
// (lane&15); K slots interleaved like A: half 0 k={0..7,16..23}, half 1 k={8..15,24..31}.
__device__ __forceinline__ v16h frag_ldB(const _Float16* tile, int ld) {
  const int lane = threadIdx.x & 31;
  const int row  = lane & 15;
  const int k0   = (lane >> 4) << 3;          // 0 or 8
  const _Float16* p = tile + row * ld + k0;
  h8 lo = *reinterpret_cast<const h8*>(p);
  h8 hi = *reinterpret_cast<const h8*>(p + 16);
  return __builtin_shufflevector(lo, hi, 0,1,2,3,4,5,6,7,8,9,10,11,12,13,14,15);
}

#define WMMA_F16(a, b, c) \
  __builtin_amdgcn_wmma_f32_16x16x32_f16(false, (a), false, (b), (short)0, (c), false, false)

// ---------------------------------------------------------------------------
// One-shot fp32 -> f16 conversion (8 elements / thread, exact grids)
// ---------------------------------------------------------------------------
__global__ __launch_bounds__(256) void cvt_f32_f16(
    const float* __restrict__ src, _Float16* __restrict__ dst, long n8)
{
  const long i = (long)blockIdx.x * blockDim.x + threadIdx.x;
  if (i >= n8) return;
  const long base = i * 8;
  const float4 a = *reinterpret_cast<const float4*>(src + base);
  const float4 b = *reinterpret_cast<const float4*>(src + base + 4);
  h8 o;
  o[0] = (_Float16)a.x; o[1] = (_Float16)a.y; o[2] = (_Float16)a.z; o[3] = (_Float16)a.w;
  o[4] = (_Float16)b.x; o[5] = (_Float16)b.y; o[6] = (_Float16)b.z; o[7] = (_Float16)b.w;
  *reinterpret_cast<h8*>(dst + base) = o;
}

// ---------------------------------------------------------------------------
// GEMM: C[M,N] = A[M,K] * Bt[N,K]^T   (f16 in, f32 acc, TOUT out)
// Block tile 64(M) x 256(N), K-step 32. 256 threads = 8 waves, wave tile 32x64.
// A/B tiles staged with global_load_async_to_lds_b128 (ASYNCcnt path).
// ---------------------------------------------------------------------------
#define GBM 64
#define GBN 256
#define GBK 32
#define GLDA (GBK + 16)   // 48 halves = 96B rows, 16B aligned

template <typename TOUT>
__global__ __launch_bounds__(256) void wmma_gemm_nt(
    const _Float16* __restrict__ A, const _Float16* __restrict__ Bt,
    TOUT* __restrict__ C, int M, int N, int K)
{
  __shared__ __align__(16) _Float16 sA[GBM][GLDA];
  __shared__ __align__(16) _Float16 sB[GBN][GLDA];

  const int tid  = threadIdx.x;
  const int wave = tid >> 5;
  const int lane = tid & 31;
  const int wm = (wave >> 2) * 32;   // 0 / 32
  const int wn = (wave & 3) * 64;    // 0 / 64 / 128 / 192
  const long m0 = (long)blockIdx.y * GBM;
  const long n0 = (long)blockIdx.x * GBN;

  // per-thread staging addresses (row parts advance with k0 each iteration)
  const int ar = tid >> 2;
  const int ac = (tid & 3) << 3;
  const unsigned ldsA = lds_off(&sA[ar][ac]);
  const unsigned ldsB = lds_off(&sB[tid][0]);

  v8f acc[2][4] = {};

  for (int k0 = 0; k0 < K; k0 += GBK) {
    __syncthreads();
    { // stage A: 64x32 halves, one async b128 per thread
      const unsigned long long ga =
          (unsigned long long)(A + (m0 + ar) * (long)K + k0 + ac);
      ASYNC_LD_B128(ldsA, ga, 0);
    }
    { // stage Bt: 256x16 halves (k0..k0+15 of each row), 2 async b128 per thread
      const unsigned long long gb =
          (unsigned long long)(Bt + (n0 + tid) * (long)K + k0);
      ASYNC_LD_B128(ldsB, gb, 0);
      ASYNC_LD_B128(ldsB, gb, 16);
    }
    WAIT_ASYNC();
    __syncthreads();

    v16h af[2], bf[4];
    #pragma unroll
    for (int mi = 0; mi < 2; ++mi) af[mi] = frag_ldA(&sA[wm + mi * 16][0], GLDA);
    #pragma unroll
    for (int ni = 0; ni < 4; ++ni) bf[ni] = frag_ldB(&sB[wn + ni * 16][0], GLDA);

    #pragma unroll
    for (int mi = 0; mi < 2; ++mi)
      #pragma unroll
      for (int ni = 0; ni < 4; ++ni)
        acc[mi][ni] = WMMA_F16(af[mi], bf[ni], acc[mi][ni]);
  }

  // C/D layout: m = r + 8*(lane>=16), n = lane&15
  const int rm = (lane >> 4) << 3;
  const int cn = lane & 15;
  #pragma unroll
  for (int mi = 0; mi < 2; ++mi)
    #pragma unroll
    for (int r = 0; r < 8; ++r) {
      const long m = m0 + wm + mi * 16 + r + rm;
      TOUT* cp = C + m * (long)N + n0 + wn + cn;
      #pragma unroll
      for (int ni = 0; ni < 4; ++ni)
        cp[ni * 16] = (TOUT)acc[mi][ni][r];
    }
}

// ---------------------------------------------------------------------------
// Flash attention, causal, f16 in / f16 out. Grid: (T/64, H, B).
// 128 threads = 4 waves; each wave owns 16 query rows x D=64.
// Key blocks of 64, online softmax. qkv16 layout: [B*T][3E].
// K staged via async global->LDS; V^T / P staged manually (permuted layouts).
// ---------------------------------------------------------------------------
#define KLD (CD + 16)    // 80 halves: sK rows [key][d]
#define VLD (64 + 16)    // 80 halves: sVt rows [d][key], sP rows [m][key]

__global__ __launch_bounds__(128) void wmma_flash_attn(
    const _Float16* __restrict__ qkv, _Float16* __restrict__ out)
{
  __shared__ __align__(16) _Float16 sK [64][KLD];
  __shared__ __align__(16) _Float16 sVt[CD][VLD];
  __shared__ __align__(16) _Float16 sP [4][16][VLD];

  const int tid  = threadIdx.x;
  const int wave = tid >> 5;
  const int lane = tid & 31;
  const int b  = blockIdx.z;
  const int h  = blockIdx.y;
  const int qb = blockIdx.x;
  const int qi0 = qb * 64 + wave * 16;

  // Q fragments (A-operand) straight from global f16
  const _Float16* qtile = qkv + ((long)(b * CT) + qi0) * C3E + h * CD;
  v16h aq[2];
  aq[0] = frag_ldA(qtile,      C3E);
  aq[1] = frag_ldA(qtile + 32, C3E);

  v8f o[4] = {};
  float Mrow[8], Lrow[8];
  #pragma unroll
  for (int r = 0; r < 8; ++r) { Mrow[r] = -1e30f; Lrow[r] = 0.0f; }

  const int rm = (lane >> 4) << 3;
  const int cn = lane & 15;
  const float scale = 0.125f;            // 1/sqrt(64)

  // staging addresses
  const int kr = tid >> 1;               // key row for K staging
  const int kc = (tid & 1) << 5;         // half-row offset (halves)
  const unsigned ldsK = lds_off(&sK[kr][kc]);
  const int kA = (tid >> 2) << 1;        // even key for V^T staging
  const int dg = (tid & 3) << 4;         // d group for V^T staging

  for (int j = 0; j <= qb; ++j) {
    const int kb = j * 64;
    __syncthreads();
    { // stage K: 64 rows x 64 halves, async copy (4 x b128 per thread)
      const unsigned long long gk = (unsigned long long)(
          qkv + ((long)(b * CT) + kb + kr) * C3E + CE + h * CD + kc);
      ASYNC_LD_B128(ldsK, gk, 0);
      ASYNC_LD_B128(ldsK, gk, 16);
      ASYNC_LD_B128(ldsK, gk, 32);
      ASYNC_LD_B128(ldsK, gk, 48);
    }
    { // stage V^T: thread handles a key pair x 16 d, packed b32 LDS stores
      const _Float16* vA = qkv + ((long)(b * CT) + kb + kA) * C3E + 2 * CE + h * CD + dg;
      const _Float16* vB = vA + C3E;
      h8 a0 = *reinterpret_cast<const h8*>(vA);
      h8 a1 = *reinterpret_cast<const h8*>(vA + 8);
      h8 b0 = *reinterpret_cast<const h8*>(vB);
      h8 b1 = *reinterpret_cast<const h8*>(vB + 8);
      #pragma unroll
      for (int i = 0; i < 8; ++i) {
        union { _Float16 hh[2]; unsigned u; } pk;
        pk.hh[0] = a0[i]; pk.hh[1] = b0[i];
        *reinterpret_cast<unsigned*>(&sVt[dg + i][kA]) = pk.u;
      }
      #pragma unroll
      for (int i = 0; i < 8; ++i) {
        union { _Float16 hh[2]; unsigned u; } pk;
        pk.hh[0] = a1[i]; pk.hh[1] = b1[i];
        *reinterpret_cast<unsigned*>(&sVt[dg + 8 + i][kA]) = pk.u;
      }
    }
    WAIT_ASYNC();
    __syncthreads();

    // S(16x64) = Q(16x64) @ K_block(64x64)^T : 8 WMMAs
    v8f s[4] = {};
    #pragma unroll
    for (int ni = 0; ni < 4; ++ni) {
      v16h bk0 = frag_ldB(&sK[ni * 16][0],  KLD);
      v16h bk1 = frag_ldB(&sK[ni * 16][32], KLD);
      s[ni] = WMMA_F16(aq[0], bk0, s[ni]);
      s[ni] = WMMA_F16(aq[1], bk1, s[ni]);
    }

    // causal mask + per-row max (rows live in one register per lane-half)
    float mx[8];
    #pragma unroll
    for (int r = 0; r < 8; ++r) {
      const int qi = qi0 + r + rm;
      float m = -1e30f;
      #pragma unroll
      for (int ni = 0; ni < 4; ++ni) {
        s[ni][r] = (kb + ni * 16 + cn <= qi) ? s[ni][r] * scale : -1e30f;
        m = fmaxf(m, s[ni][r]);
      }
      m = fmaxf(m, __shfl_xor(m, 1, 32));
      m = fmaxf(m, __shfl_xor(m, 2, 32));
      m = fmaxf(m, __shfl_xor(m, 4, 32));
      m = fmaxf(m, __shfl_xor(m, 8, 32));
      mx[r] = m;
    }
    // online softmax update (in place on s[])
    #pragma unroll
    for (int r = 0; r < 8; ++r) {
      const float mnew = fmaxf(Mrow[r], mx[r]);
      const float corr = __expf(Mrow[r] - mnew);
      Mrow[r] = mnew;
      float rs = 0.0f;
      #pragma unroll
      for (int ni = 0; ni < 4; ++ni) {
        s[ni][r] = __expf(s[ni][r] - mnew);
        rs += s[ni][r];
      }
      rs += __shfl_xor(rs, 1, 32);
      rs += __shfl_xor(rs, 2, 32);
      rs += __shfl_xor(rs, 4, 32);
      rs += __shfl_xor(rs, 8, 32);
      Lrow[r] = Lrow[r] * corr + rs;
      #pragma unroll
      for (int g = 0; g < 4; ++g) o[g][r] *= corr;
    }

    // re-layout P (C-layout) -> A-operand fragments via LDS
    #pragma unroll
    for (int r = 0; r < 8; ++r)
      #pragma unroll
      for (int ni = 0; ni < 4; ++ni)
        sP[wave][r + rm][ni * 16 + cn] = (_Float16)s[ni][r];
    __syncthreads();
    const v16h pf0 = frag_ldA(&sP[wave][0][0],  VLD);   // keys 0..31
    const v16h pf1 = frag_ldA(&sP[wave][0][32], VLD);   // keys 32..63

    // O(16x64) += P(16x64) @ V(64x64) : 8 WMMAs
    #pragma unroll
    for (int g = 0; g < 4; ++g) {
      v16h vb0 = frag_ldB(&sVt[g * 16][0],  VLD);
      v16h vb1 = frag_ldB(&sVt[g * 16][32], VLD);
      o[g] = WMMA_F16(pf0, vb0, o[g]);
      o[g] = WMMA_F16(pf1, vb1, o[g]);
    }
  }

  // normalize and write f16 [B*T][E] at column h*D
  #pragma unroll
  for (int r = 0; r < 8; ++r) {
    const float inv = 1.0f / Lrow[r];
    const long mrow = (long)(b * CT) + qi0 + r + rm;
    _Float16* op = out + mrow * CE + h * CD;
    #pragma unroll
    for (int g = 0; g < 4; ++g)
      op[g * 16 + cn] = (_Float16)(o[g][r] * inv);
  }
}

// ---------------------------------------------------------------------------
extern "C" void kernel_launch(void* const* d_in, const int* in_sizes, int n_in,
                              void* d_out, int out_size, void* d_ws, size_t ws_size,
                              hipStream_t stream) {
  (void)in_sizes; (void)n_in; (void)out_size; (void)ws_size;
  const float* x      = (const float*)d_in[0];   // [B,T,E]
  const float* w_qkv  = (const float*)d_in[1];   // [3E,E]
  const float* w_proj = (const float*)d_in[2];   // [E,E]
  float* outp = (float*)d_out;                   // [B,T,E]

  const long M = (long)CB * CT;                  // 4096

  _Float16* x16     = (_Float16*)d_ws;                     // M*E
  _Float16* wqkv16  = x16     + M * CE;                    // 3E*E
  _Float16* wproj16 = wqkv16  + (long)C3E * CE;            // E*E
  _Float16* qkv16   = wproj16 + (long)CE * CE;             // M*3E
  _Float16* attn16  = qkv16   + M * C3E;                   // M*E

  // 0) fp32 -> f16, once per tensor
  {
    const long n8x = M * CE / 8, n8q = (long)C3E * CE / 8, n8p = (long)CE * CE / 8;
    cvt_f32_f16<<<dim3(n8x / 256), 256, 0, stream>>>(x,      x16,     n8x);
    cvt_f32_f16<<<dim3(n8q / 256), 256, 0, stream>>>(w_qkv,  wqkv16,  n8q);
    cvt_f32_f16<<<dim3(n8p / 256), 256, 0, stream>>>(w_proj, wproj16, n8p);
  }

  // 1) qkv16 = x16 @ wqkv16^T   (M=4096, N=3072, K=1024), f16 out
  dim3 g1(C3E / GBN, M / GBM);
  wmma_gemm_nt<_Float16><<<g1, 256, 0, stream>>>(x16, wqkv16, qkv16, M, C3E, CE);

  // 2) fused causal flash attention -> attn16 [B*T, E]
  dim3 g2(CT / 64, CH, CB);
  wmma_flash_attn<<<g2, 128, 0, stream>>>(qkv16, attn16);

  // 3) out = attn16 @ wproj16^T  (M=4096, N=1024, K=1024), f32 out
  dim3 g3(CE / GBN, M / GBM);
  wmma_gemm_nt<float><<<g3, 256, 0, stream>>>(attn16, wproj16, outp, M, CE, CE);
}
